// GRUFeatureExtractor_71751723647460
// MI455X (gfx1250) — compile-verified, matrix operand order checked by reading, not run on hardware
//
#include <hip/hip_runtime.h>
#include <hip/hip_bf16.h>
#include <math.h>

// ---------------- model dims ----------------
#define Bn   4096
#define SEQn 10
#define Hn   512
#define En   1024           // 2*H
#define Gn   1536           // 3*H
// GEMM tiling
#define BM 128
#define BN 128
#define BK 32

typedef __attribute__((ext_vector_type(16))) __bf16 bfx16;
typedef __attribute__((ext_vector_type(8)))  __bf16 bfx8;
typedef __attribute__((ext_vector_type(8)))  float  fx8;
typedef __attribute__((ext_vector_type(4)))  unsigned int u32x4;
typedef __attribute__((ext_vector_type(8)))  int          i32x8;
typedef __attribute__((ext_vector_type(4)))  int          i32x4;

struct __align__(8) US4 { unsigned short x, y, z, w; };

__device__ __forceinline__ unsigned short f2bf_u(float f) {
  union { float f; unsigned u; } x; x.f = f;
  unsigned r = x.u + 0x7FFFu + ((x.u >> 16) & 1u);   // round-to-nearest-even
  return (unsigned short)(r >> 16);
}

__device__ __forceinline__ float sigmoidf_(float x) { return 1.0f / (1.0f + expf(-x)); }

// ---- Tensor Data Mover: 2D tile load (BN rows x 64B) of bf16 weights into LDS ----
// D# group0: count=1 | lds_addr | global_addr(57b) | type=2("image")
// D# group1: data_size=2(4B), tensor_dim0=16 dwords, tensor_dim1=BN rows,
//            tile_dim0=16 dwords, tile_dim1=BN, tensor_dim0_stride=K/2 dwords.
// Groups 2/3 zero (2D tensor: tile_dim2+ unused).  6-arg builtin form (clang-23 lane).
__device__ __forceinline__ void tdm_load_Btile(const unsigned short* gsrc,
                                               unsigned ldsOff, int K) {
  unsigned long long ga = (unsigned long long)(size_t)gsrc;
  u32x4 g0;
  g0[0] = 1u;                                           // count=1, user descriptor
  g0[1] = ldsOff;                                       // LDS byte address
  g0[2] = (unsigned)(ga & 0xFFFFFFFFull);               // global_addr[31:0]
  g0[3] = (unsigned)((ga >> 32) & 0x01FFFFFFull) | (2u << 30);  // addr[56:32] | type=2

  const unsigned td0     = 16u;                          // row length in dwords (32 bf16)
  const unsigned rows    = (unsigned)BN;                 // 128 rows
  const unsigned strideD = (unsigned)(K >> 1);           // row stride in dwords
  i32x8 g1;
  g1[0] = (int)(2u << 16);                               // workgroup_mask=0, data_size=4B
  g1[1] = (int)((td0 & 0xFFFFu) << 16);                  // abar=0 | tensor_dim0[15:0]
  g1[2] = (int)(((td0 >> 16) & 0xFFFFu) | ((rows & 0xFFFFu) << 16)); // dim0 hi | dim1 lo
  g1[3] = (int)(((rows >> 16) & 0xFFFFu) | (td0 << 16)); // dim1 hi | tile_dim0
  g1[4] = (int)(rows & 0xFFFFu);                         // tile_dim1 | tile_dim2=0
  g1[5] = (int)strideD;                                  // tensor_dim0_stride[31:0]
  g1[6] = 0;                                             // stride hi | dim1_stride lo
  g1[7] = 0;
  i32x4 z4 = {0, 0, 0, 0};
  i32x8 z8 = {0, 0, 0, 0, 0, 0, 0, 0};
  __builtin_amdgcn_tensor_load_to_lds(g0, g1, z4, z4, z8, 0);
}

// ---------------- weight cast kernels ----------------
__global__ void k_cast_bf16(const float* __restrict__ src, unsigned short* __restrict__ dst, int n) {
  int i = blockIdx.x * 256 + threadIdx.x;
  if (i < n) dst[i] = f2bf_u(src[i]);
}

// Wroll[g][j] = W[g][(j+1) % 512]  (absorbs jnp.roll on the activation into the weight)
__global__ void k_cast_roll512(const float* __restrict__ src, unsigned short* __restrict__ dst, int n) {
  int i = blockIdx.x * 256 + threadIdx.x;
  if (i < n) {
    int j = i & 511; int base = i - j;
    dst[i] = f2bf_u(src[base + ((j + 1) & 511)]);
  }
}

__global__ void k_fill0(float* __restrict__ p, int n) {
  int i = blockIdx.x * 256 + threadIdx.x;
  if (i < n) p[i] = 0.0f;
}

// ---------------- WMMA GEMM: C(M,N) = epi(A(M,K) @ Wb(N,K)^T + bias) ----------------
// A fp32 (converted to bf16 while staging to LDS by VALU), Wb bf16 row-major (N,K) moved
// by the Tensor Data Mover, C fp32. Double-buffered LDS ping-pong: while tile k feeds the
// WMMA pipes, tile k+1 is staged (A by VALU, B by TDM) into the other buffer.
// Block: 256 threads = 8 waves; wave (wm 0..3, wn 0..1) computes 32x64 via 2x4 WMMA frags.
template<int EPI>
__global__ __launch_bounds__(256)
void k_gemm(const float* __restrict__ A, int lda,
            const unsigned short* __restrict__ Wb,
            const float* __restrict__ bias,
            float* __restrict__ C, int N, int K) {
  __shared__ unsigned short sA[2][BM * BK];
  __shared__ unsigned short sB[2][BN * BK];

  const int tid   = threadIdx.x;
  const int rowA0 = blockIdx.y * BM;
  const int colN0 = blockIdx.x * BN;
  const int wave  = tid >> 5, lane = tid & 31;
  const int wm    = wave >> 1, wn = wave & 1;
  const int half  = lane >> 4, l16 = lane & 15;
  const int nk    = K / BK;

  fx8 acc[2][4];
#pragma unroll
  for (int mi = 0; mi < 2; ++mi)
#pragma unroll
    for (int nj = 0; nj < 4; ++nj)
#pragma unroll
      for (int e = 0; e < 8; ++e) acc[mi][nj][e] = 0.0f;

  auto stageA = [&](int k0, int buf) {
    for (int u = tid; u < (BM * BK) / 4; u += 256) {
      int r = u >> 3; int c = (u & 7) << 2;
      const float4 v = *(const float4*)(A + (size_t)(rowA0 + r) * lda + (k0 + c));
      US4 h4; h4.x = f2bf_u(v.x); h4.y = f2bf_u(v.y); h4.z = f2bf_u(v.z); h4.w = f2bf_u(v.w);
      *(US4*)&sA[buf][r * BK + c] = h4;
    }
  };
  auto stageB = [&](int k0, int buf) {
    if (tid < 32)   // one TDM issue per block (scalar op, EXEC ignored)
      tdm_load_Btile(Wb + (size_t)colN0 * K + k0,
                     (unsigned)(size_t)&sB[buf][0], K);
  };

  // prologue: stage tile 0 into buffer 0
  stageA(0, 0);
  stageB(0, 0);

  for (int kt = 0; kt < nk; ++kt) {
    const int cur = kt & 1;
    __builtin_amdgcn_s_wait_tensorcnt(0);   // wave0: TDM into sB[cur] complete
    __syncthreads();                        // publish tile cur; all reads of cur^1 retired
    if (kt + 1 < nk) {
      stageA((kt + 1) * BK, cur ^ 1);
      stageB((kt + 1) * BK, cur ^ 1);
      if (kt + 2 < nk && tid < 128)
        __builtin_prefetch(A + (size_t)(rowA0 + tid) * lda + (kt + 2) * BK, 0, 3);
    }

    // fragment loads per ISA 16-bit layout: lanes 0-15 -> K[0..7],[16..23]; 16-31 -> K[8..15],[24..31]
    bfx16 afrag[2], bfrag[4];
#pragma unroll
    for (int mi = 0; mi < 2; ++mi) {
      int row = wm * 32 + mi * 16 + l16;
      const bfx8 lo = *(const bfx8*)&sA[cur][row * BK + half * 8];
      const bfx8 hi = *(const bfx8*)&sA[cur][row * BK + 16 + half * 8];
#pragma unroll
      for (int e = 0; e < 8; ++e) { afrag[mi][e] = lo[e]; afrag[mi][8 + e] = hi[e]; }
    }
#pragma unroll
    for (int nj = 0; nj < 4; ++nj) {
      int row = wn * 64 + nj * 16 + l16;      // B column n == W row n (contiguous over K)
      const bfx8 lo = *(const bfx8*)&sB[cur][row * BK + half * 8];
      const bfx8 hi = *(const bfx8*)&sB[cur][row * BK + 16 + half * 8];
#pragma unroll
      for (int e = 0; e < 8; ++e) { bfrag[nj][e] = lo[e]; bfrag[nj][8 + e] = hi[e]; }
    }
#pragma unroll
    for (int mi = 0; mi < 2; ++mi)
#pragma unroll
      for (int nj = 0; nj < 4; ++nj)
        acc[mi][nj] = __builtin_amdgcn_wmma_f32_16x16x32_bf16(
            false, afrag[mi], false, bfrag[nj], (short)0, acc[mi][nj], false, false);
  }

  // epilogue: C VGPR e -> row (e + 8*half), col = lane%16
#pragma unroll
  for (int mi = 0; mi < 2; ++mi) {
#pragma unroll
    for (int nj = 0; nj < 4; ++nj) {
      int col = colN0 + wn * 64 + nj * 16 + l16;
      float bv = bias ? bias[col] : 0.0f;
#pragma unroll
      for (int e = 0; e < 8; ++e) {
        int row = rowA0 + wm * 32 + mi * 16 + half * 8 + e;
        float v = acc[mi][nj][e] + bv;
        if (EPI == 1) v = tanhf(v);
        C[(size_t)row * N + col] = v;
      }
    }
  }
}

// ---------------- GRU gate kernels ----------------
// Layer 0: gi_t = alpha*P + beta*R + bih (phase factorization); gh already contains bhh.
__global__ void k_gru_gate_l0(const float* __restrict__ P, const float* __restrict__ R,
                              const float* __restrict__ gh, const float* __restrict__ bih,
                              float* __restrict__ h, float* __restrict__ out,
                              float alpha, float beta) {
  int idx = blockIdx.x * 256 + threadIdx.x;
  if (idx >= Bn * Hn) return;
  int b = idx >> 9; int j = idx & 511;
  size_t r3 = (size_t)b * Gn;
  float ir  = alpha * P[r3 + j]        + beta * R[r3 + j]        + bih[j];
  float iz  = alpha * P[r3 + 512 + j]  + beta * R[r3 + 512 + j]  + bih[512 + j];
  float in_ = alpha * P[r3 + 1024 + j] + beta * R[r3 + 1024 + j] + bih[1024 + j];
  float rg = sigmoidf_(ir + gh[r3 + j]);
  float zg = sigmoidf_(iz + gh[r3 + 512 + j]);
  float ng = tanhf(in_ + rg * gh[r3 + 1024 + j]);
  float hp = h[(size_t)b * Hn + j];
  float hn = (1.0f - zg) * ng + zg * hp;
  h[(size_t)b * Hn + j] = hn;
  out[(size_t)b * (SEQn * En) + j] = hn;   // out pre-offset by t*E + d*H
}

// Layer 1: gi (bih included via GEMM bias) indexed at row b*SEQ+t (pointer pre-offset by t*G).
__global__ void k_gru_gate_l1(const float* __restrict__ gi, const float* __restrict__ gh,
                              float* __restrict__ h, float* __restrict__ out) {
  int idx = blockIdx.x * 256 + threadIdx.x;
  if (idx >= Bn * Hn) return;
  int b = idx >> 9; int j = idx & 511;
  size_t gr = (size_t)b * (SEQn * Gn);
  size_t r3 = (size_t)b * Gn;
  float rg = sigmoidf_(gi[gr + j]        + gh[r3 + j]);
  float zg = sigmoidf_(gi[gr + 512 + j]  + gh[r3 + 512 + j]);
  float ng = tanhf(    gi[gr + 1024 + j] + rg * gh[r3 + 1024 + j]);
  float hp = h[(size_t)b * Hn + j];
  float hn = (1.0f - zg) * ng + zg * hp;
  h[(size_t)b * Hn + j] = hn;
  out[(size_t)b * (SEQn * En) + j] = hn;
}

// ---------------- attention (S=10, HD=128): one block per (b, head) ----------------
__global__ __launch_bounds__(128)
void k_attn(const float* __restrict__ qkv, float* __restrict__ attn_o) {
  int bh = blockIdx.x; int b = bh >> 3; int h = bh & 7;
  __shared__ float sQ[SEQn][128], sK[SEQn][128], sV[SEQn][128];
  __shared__ float sS[SEQn][12];
  int tid = threadIdx.x;
  for (int t = 0; t < SEQn; ++t) {
    size_t base = (size_t)(b * SEQn + t) * 3072 + h * 128 + tid;
    sQ[t][tid] = qkv[base];
    sK[t][tid] = qkv[base + 1024];
    sV[t][tid] = qkv[base + 2048];
  }
  __syncthreads();
  if (tid < SEQn * SEQn) {
    int i = tid / SEQn, j = tid % SEQn;
    float acc = 0.0f;
    for (int d2 = 0; d2 < 128; ++d2) acc += sQ[i][d2] * sK[j][d2];
    sS[i][j] = acc * 0.08838834764831845f;   // 1/sqrt(128)
  }
  __syncthreads();
  if (tid < SEQn) {
    float m = -1e30f;
    for (int j = 0; j < SEQn; ++j) m = fmaxf(m, sS[tid][j]);
    float s = 0.0f;
    for (int j = 0; j < SEQn; ++j) { float e = expf(sS[tid][j] - m); sS[tid][j] = e; s += e; }
    float inv = 1.0f / s;
    for (int j = 0; j < SEQn; ++j) sS[tid][j] *= inv;
  }
  __syncthreads();
  for (int i = 0; i < SEQn; ++i) {
    float acc = 0.0f;
#pragma unroll
    for (int j = 0; j < SEQn; ++j) acc += sS[i][j] * sV[j][tid];
    attn_o[(size_t)(b * SEQn + i) * En + h * 128 + tid] = acc;
  }
}

// ---------------- mean over t of (gru_out + attn_out) ----------------
__global__ void k_aggregate(const float* __restrict__ a, const float* __restrict__ c,
                            float* __restrict__ outp, int n) {
  int i = blockIdx.x * 256 + threadIdx.x;
  if (i >= n) return;
  int b = i >> 10; int e = i & 1023;
  size_t base = (size_t)b * (SEQn * En) + e;
  float s = 0.0f;
#pragma unroll
  for (int t = 0; t < SEQn; ++t) s += a[base + (size_t)t * En] + c[base + (size_t)t * En];
  outp[i] = s * 0.1f;
}

// ---------------- host orchestration ----------------
extern "C" void kernel_launch(void* const* d_in, const int* in_sizes, int n_in,
                              void* d_out, int out_size, void* d_ws, size_t ws_size,
                              hipStream_t stream) {
  (void)in_sizes; (void)n_in; (void)out_size; (void)ws_size;
  const float* x    = (const float*)d_in[0];
  const float* Wp   = (const float*)d_in[1];
  const float* bp   = (const float*)d_in[2];
  const float* Wih0 = (const float*)d_in[3];
  const float* Whh0 = (const float*)d_in[4];
  const float* bih0 = (const float*)d_in[5];
  const float* bhh0 = (const float*)d_in[6];
  const float* Wih1 = (const float*)d_in[7];
  const float* Whh1 = (const float*)d_in[8];
  const float* bih1 = (const float*)d_in[9];
  const float* bhh1 = (const float*)d_in[10];
  const float* inW  = (const float*)d_in[11];
  const float* inB  = (const float*)d_in[12];
  const float* outW = (const float*)d_in[13];
  const float* outB = (const float*)d_in[14];
  const float* W1   = (const float*)d_in[15];
  const float* b1   = (const float*)d_in[16];
  const float* W2   = (const float*)d_in[17];
  const float* b2   = (const float*)d_in[18];
  float* outp = (float*)d_out;

  char* base = (char*)d_ws; size_t off = 0;
  auto alloc = [&](size_t bytes) -> void* {
    off = (off + 255) & ~(size_t)255; void* p = base + off; off += bytes; return p;
  };

  // bf16 weight copies
  unsigned short* wp_bf    = (unsigned short*)alloc((size_t)512 * 512 * 2);
  unsigned short* wih0_bf  = (unsigned short*)alloc((size_t)2 * Gn * Hn * 2);
  unsigned short* wih0r_bf = (unsigned short*)alloc((size_t)2 * Gn * Hn * 2);
  unsigned short* whh0_bf  = (unsigned short*)alloc((size_t)2 * Gn * Hn * 2);
  unsigned short* wih1_bf  = (unsigned short*)alloc((size_t)2 * Gn * En * 2);
  unsigned short* whh1_bf  = (unsigned short*)alloc((size_t)2 * Gn * Hn * 2);
  unsigned short* inw_bf   = (unsigned short*)alloc((size_t)3072 * 1024 * 2);
  unsigned short* outw_bf  = (unsigned short*)alloc((size_t)1024 * 1024 * 2);
  unsigned short* w1_bf    = (unsigned short*)alloc((size_t)128 * 1024 * 2);
  unsigned short* w2_bf    = (unsigned short*)alloc((size_t)256 * 128 * 2);
  // fp32 activations
  float* x_proj = (float*)alloc((size_t)Bn * Hn * 4);
  float* Pbuf   = (float*)alloc((size_t)2 * Bn * Gn * 4);
  float* Rbuf   = (float*)alloc((size_t)2 * Bn * Gn * 4);
  float* ghbuf  = (float*)alloc((size_t)2 * Bn * Gn * 4);
  float* hcur   = (float*)alloc((size_t)2 * Bn * Hn * 4);
  float* h1     = (float*)alloc((size_t)Bn * SEQn * En * 4);
  float* gi1    = (float*)alloc((size_t)2 * Bn * SEQn * Gn * 4);
  float* gruout = (float*)alloc((size_t)Bn * SEQn * En * 4);
  float* qkv    = gi1;   // alias: gi1 dead before qkv is produced (same size)
  float* attn_o = h1;    // alias: h1 dead after gi1 GEMMs (same size)
  float* attnout= (float*)alloc((size_t)Bn * SEQn * En * 4);
  float* agg    = (float*)alloc((size_t)Bn * En * 4);
  float* hid    = (float*)alloc((size_t)Bn * 128 * 4);

  auto cast = [&](const float* s, unsigned short* dst, int n) {
    k_cast_bf16<<<(n + 255) / 256, 256, 0, stream>>>(s, dst, n);
  };
  cast(Wp, wp_bf, 512 * 512);
  cast(Wih0, wih0_bf, 2 * Gn * Hn);
  k_cast_roll512<<<(2 * Gn * Hn + 255) / 256, 256, 0, stream>>>(Wih0, wih0r_bf, 2 * Gn * Hn);
  cast(Whh0, whh0_bf, 2 * Gn * Hn);
  cast(Wih1, wih1_bf, 2 * Gn * En);
  cast(Whh1, whh1_bf, 2 * Gn * Hn);
  cast(inW, inw_bf, 3072 * 1024);
  cast(outW, outw_bf, 1024 * 1024);
  cast(W1, w1_bf, 128 * 1024);
  cast(W2, w2_bf, 256 * 128);

  auto gemm = [&](const float* A, int lda, const unsigned short* Wb, const float* bias,
                  float* C, int M, int N, int K, int epi) {
    dim3 grid(N / BN, M / BM);
    if (epi) k_gemm<1><<<grid, 256, 0, stream>>>(A, lda, Wb, bias, C, N, K);
    else     k_gemm<0><<<grid, 256, 0, stream>>>(A, lda, Wb, bias, C, N, K);
  };

  // x_proj = x @ Wp^T + bp
  gemm(x, 512, wp_bf, bp, x_proj, Bn, 512, 512, 0);

  // phase factorization: P_d = x_proj @ Wih0[d]^T ; R_d = x_proj @ roll(Wih0[d])^T
  for (int d = 0; d < 2; ++d) {
    gemm(x_proj, 512, wih0_bf  + (size_t)d * Gn * Hn, nullptr, Pbuf + (size_t)d * Bn * Gn, Bn, Gn, Hn, 0);
    gemm(x_proj, 512, wih0r_bf + (size_t)d * Gn * Hn, nullptr, Rbuf + (size_t)d * Bn * Gn, Bn, Gn, Hn, 0);
  }

  const double kPI = 3.14159265358979323846;
  const int nh = 2 * Bn * Hn;
  const int gateGrid = (Bn * Hn + 255) / 256;

  // layer 0 scan (fwd: t = s; bwd: t = SEQ-1-s, stored at t)
  k_fill0<<<(nh + 255) / 256, 256, 0, stream>>>(hcur, nh);
  for (int s = 0; s < SEQn; ++s) {
    for (int d = 0; d < 2; ++d) {
      int t = d ? (SEQn - 1 - s) : s;
      double ph = 2.0 * kPI * t / SEQn;
      float alpha = (float)(1.0 + 0.1 * cos(ph));
      float beta  = (float)(0.05 * sin(ph));
      gemm(hcur + (size_t)d * Bn * Hn, Hn, whh0_bf + (size_t)d * Gn * Hn, bhh0 + d * Gn,
           ghbuf + (size_t)d * Bn * Gn, Bn, Gn, Hn, 0);
      k_gru_gate_l0<<<gateGrid, 256, 0, stream>>>(
          Pbuf + (size_t)d * Bn * Gn, Rbuf + (size_t)d * Bn * Gn, ghbuf + (size_t)d * Bn * Gn,
          bih0 + d * Gn, hcur + (size_t)d * Bn * Hn,
          h1 + (size_t)t * En + (size_t)d * Hn, alpha, beta);
    }
  }

  // layer 1 input gates: gi1_d = h1 @ Wih1[d]^T + bih1[d]
  for (int d = 0; d < 2; ++d)
    gemm(h1, En, wih1_bf + (size_t)d * Gn * En, bih1 + d * Gn,
         gi1 + (size_t)d * Bn * SEQn * Gn, Bn * SEQn, Gn, En, 0);

  // layer 1 scan
  k_fill0<<<(nh + 255) / 256, 256, 0, stream>>>(hcur, nh);
  for (int s = 0; s < SEQn; ++s) {
    for (int d = 0; d < 2; ++d) {
      int t = d ? (SEQn - 1 - s) : s;
      gemm(hcur + (size_t)d * Bn * Hn, Hn, whh1_bf + (size_t)d * Gn * Hn, bhh1 + d * Gn,
           ghbuf + (size_t)d * Bn * Gn, Bn, Gn, Hn, 0);
      k_gru_gate_l1<<<gateGrid, 256, 0, stream>>>(
          gi1 + (size_t)d * Bn * SEQn * Gn + (size_t)t * Gn, ghbuf + (size_t)d * Bn * Gn,
          hcur + (size_t)d * Bn * Hn, gruout + (size_t)t * En + (size_t)d * Hn);
    }
  }

  // attention
  gemm(gruout, En, inw_bf, inB, qkv, Bn * SEQn, 3072, En, 0);
  k_attn<<<Bn * 8, 128, 0, stream>>>(qkv, attn_o);
  gemm(attn_o, En, outw_bf, outB, attnout, Bn * SEQn, En, En, 0);

  // aggregate + MLP head
  k_aggregate<<<(Bn * En + 255) / 256, 256, 0, stream>>>(gruout, attnout, agg, Bn * En);
  gemm(agg, En, w1_bf, b1, hid, Bn, 128, En, 1);     // tanh epilogue
  gemm(hid, 128, w2_bf, b2, outp, Bn, 256, 128, 0);  // final output (4096 x 256)
}
